// postProcessing_46127948759498
// MI455X (gfx1250) — compile-verified, matrix operand order checked by reading, not run on hardware
//
#include <hip/hip_runtime.h>
#include <hip/hip_bf16.h>

// ---------------- problem constants ----------------
#define NB        8
#define NCLS      80
#define KTOP      1000
#define NPOS      21824      // 128^2 + 64^2 + 32^2 + 16^2 + 8^2
#define CANDPAD   1024       // padded candidate count (>= KTOP, mult of 32)
#define SCORE_TH  0.05f
#define IOU_TH    0.5f

typedef __attribute__((ext_vector_type(16))) _Float16 v16h;
typedef __attribute__((ext_vector_type(8)))  float    v8f;
typedef unsigned long long u64;

// =====================================================================
// Stage A: per-(image,position) class max/argmax + box decode.
// Emits score, kind(float), monotonic u32 key, and float4 box.
// =====================================================================
__global__ void decode_kernel(
    const float* __restrict__ c0, const float* __restrict__ c1,
    const float* __restrict__ c2, const float* __restrict__ c3,
    const float* __restrict__ c4,
    const float* __restrict__ r0, const float* __restrict__ r1,
    const float* __restrict__ r2, const float* __restrict__ r3,
    const float* __restrict__ r4,
    float* __restrict__ score, float* __restrict__ kindA,
    unsigned* __restrict__ key, float* __restrict__ box)
{
    int g = blockIdx.x * blockDim.x + threadIdx.x;
    if (g >= NB * NPOS) return;
    int b = g / NPOS;
    int p = g - b * NPOS;

    const float* cl; const float* rg; int W, HW, off, st;
    if      (p < 16384) { cl = c0; rg = r0; W = 128; HW = 16384; off = 0;     st = 8;   }
    else if (p < 20480) { cl = c1; rg = r1; W = 64;  HW = 4096;  off = 16384; st = 16;  }
    else if (p < 21504) { cl = c2; rg = r2; W = 32;  HW = 1024;  off = 20480; st = 32;  }
    else if (p < 21760) { cl = c3; rg = r3; W = 16;  HW = 256;   off = 21504; st = 64;  }
    else                { cl = c4; rg = r4; W = 8;   HW = 64;    off = 21760; st = 128; }

    int hw = p - off;
    int y  = hw / W;
    int x  = hw - y * W;

    // class max / argmax (first occurrence wins, matching jnp.argmax)
    const float* cp = cl + (size_t)b * NCLS * HW + hw;
    float best = cp[0];
    int   bk   = 0;
    #pragma unroll 4
    for (int c = 1; c < NCLS; ++c) {
        if ((c & 15) == 1 && c + 16 < NCLS)
            __builtin_prefetch(cp + (size_t)(c + 16) * HW, 0, 0);  // global_prefetch_b8
        float v = cp[(size_t)c * HW];
        if (v > best) { best = v; bk = c; }
    }
    float s = (best > SCORE_TH) ? best : 0.0f;

    // box decode
    const float* rp = rg + (size_t)b * 4 * HW + hw;
    float fst = (float)st;
    float e0 = rp[0]            * fst;
    float e1 = rp[(size_t)HW]   * fst;
    float e2 = rp[(size_t)2*HW] * fst;
    float e3 = rp[(size_t)3*HW] * fst;
    float cx = (x + 0.5f) * fst;
    float cy = (y + 0.5f) * fst;

    score[g] = s;
    kindA[g] = (float)bk;
    key[g]   = __float_as_uint(s);   // s >= 0 -> bits are order-preserving
    ((float4*)box)[g] = make_float4(cx - e0, cy - e1, cx + e2, cy + e3);
}

// =====================================================================
// Stage B: exact top-K *set* selection per image via MSB-first 8-bit
// radix select; ties at the threshold filled in ascending index order
// (jax.lax.top_k stable semantics). Candidates padded to 1024.
// Emits, per candidate slot: composite 64-bit ordering key
//   ckey = (score_bits << 32) | ~idx   (sentinel = 0)
// so "j outranks i"  <=>  ckey[j] > ckey[i]   (single u64 compare).
// =====================================================================
__global__ void __launch_bounds__(256) select_topk_kernel(
    const unsigned* __restrict__ key, const float* __restrict__ score,
    float* __restrict__ cscore, int* __restrict__ cidx, u64* __restrict__ ckey)
{
    int b   = blockIdx.x;
    int tid = threadIdx.x;                 // 256 threads
    const unsigned* kk = key   + (size_t)b * NPOS;
    const float*    ss = score + (size_t)b * NPOS;

    __shared__ unsigned bins[256];
    __shared__ unsigned scn[256];
    __shared__ unsigned sh_prefix, sh_kr, sh_run;

    unsigned prefix = 0, kr = KTOP;
    for (int shift = 24; shift >= 0; shift -= 8) {
        bins[tid] = 0;
        __syncthreads();
        for (int p = tid; p < NPOS; p += 256) {
            unsigned k = kk[p];
            if (shift == 24 || ((k ^ prefix) >> (shift + 8)) == 0)
                atomicAdd(&bins[(k >> shift) & 255u], 1u);
        }
        __syncthreads();
        if (tid == 0) {
            unsigned cum = 0; int sel = 0; unsigned nkr = 0;
            for (int d = 255; d >= 0; --d) {
                unsigned c = bins[d];
                if (cum + c >= kr) { sel = d; nkr = kr - cum; break; }
                cum += c;
            }
            sh_prefix = prefix | ((unsigned)sel << shift);
            sh_kr     = nkr;
        }
        __syncthreads();
        prefix = sh_prefix;
        kr     = sh_kr;
        __syncthreads();
    }
    const unsigned T        = prefix;       // K-th largest key value
    const unsigned count_gt = KTOP - kr;    // #keys strictly greater than T

    float* cs = cscore + (size_t)b * CANDPAD;
    int*   ci = cidx   + (size_t)b * CANDPAD;
    u64*   ck = ckey   + (size_t)b * CANDPAD;

    // pass 1: keys > T -> slots [0, count_gt) in arbitrary order
    if (tid == 0) sh_run = 0;
    __syncthreads();
    for (int p = tid; p < NPOS; p += 256) {
        unsigned k = kk[p];
        if (k > T) {
            unsigned s = atomicAdd(&sh_run, 1u);
            ci[s] = p; cs[s] = ss[p];
            ck[s] = ((u64)k << 32) | (u64)(~(unsigned)p);
        }
    }
    __syncthreads();
    // pass 2: keys == T, smallest kr indices, via block prefix scan
    if (tid == 0) sh_run = 0;
    __syncthreads();
    for (int base = 0; base < NPOS; base += 256) {
        int p = base + tid;
        unsigned flag = (p < NPOS && kk[p] == T) ? 1u : 0u;
        unsigned run  = sh_run;
        scn[tid] = flag;
        __syncthreads();
        for (int ofs = 1; ofs < 256; ofs <<= 1) {
            unsigned v = (tid >= ofs) ? scn[tid - ofs] : 0u;
            __syncthreads();
            scn[tid] += v;
            __syncthreads();
        }
        unsigned pos = run + scn[tid] - flag;   // exclusive prefix
        if (flag && pos < kr) {
            int s = count_gt + pos;
            ci[s] = p; cs[s] = ss[p];
            ck[s] = ((u64)T << 32) | (u64)(~(unsigned)p);
        }
        unsigned tot = scn[255];
        __syncthreads();
        if (tid == 0) sh_run = run + tot;
        __syncthreads();
    }
    // pad sentinels: ckey = 0 is strictly below every real candidate and
    // equal among sentinels -> each sentinel ranks exactly KTOP (skipped).
    for (int s = KTOP + tid; s < CANDPAD; s += 256) {
        ci[s] = 0x7FFFFFFF; cs[s] = -1.0f; ck[s] = 0ull;
    }
}

// =====================================================================
// Stage C: exact rank of each candidate among the padded 1024 via
// WMMA pairwise counting:  rank[i] = #{ j : ckey[j] > ckey[i] }.
// A = f16 0/1 indicators (16 rows x 32 cols) built with one branchless
// u64 compare per element; B = all-ones; C/D = f32 exact counts.
// One wave handles 16 rows. Scatter boxes/kind/score into out[b][rank].
// =====================================================================
__global__ void __launch_bounds__(32) rank_sort_kernel(
    const float* __restrict__ cscore, const int* __restrict__ cidx,
    const u64* __restrict__ ckey,
    const float* __restrict__ box, const float* __restrict__ kindA,
    float* __restrict__ out)
{
    int b   = blockIdx.x >> 6;       // 64 row-blocks per image
    int rb  = blockIdx.x & 63;
    int tid = threadIdx.x;           // 32 (one wave)

    __shared__ u64   s_key[CANDPAD];
    __shared__ float s_sc[CANDPAD];
    __shared__ int   s_id[CANDPAD];
    __shared__ int   s_rank[16];

    for (int i = tid; i < CANDPAD; i += 32) {
        s_key[i] = ckey  [(size_t)b * CANDPAD + i];
        s_sc[i]  = cscore[(size_t)b * CANDPAD + i];
        s_id[i]  = cidx  [(size_t)b * CANDPAD + i];
    }
    __syncthreads();

    // A-matrix layout: lanes L and L+16 both hold row M = L%16; together
    // they cover all 32 K slots. B is all ones, so which 16 of the 32
    // j-columns each half-lane packs is irrelevant to the row sum.
    u64 ki = s_key[rb * 16 + (tid & 15)];

    v16h ones;
    #pragma unroll
    for (int t = 0; t < 16; ++t) ones[t] = (_Float16)1.0f;

    v8f acc = {};
    #pragma unroll 2
    for (int j0 = 0; j0 < CANDPAD; j0 += 32) {
        int jb = j0 + ((tid >> 4) << 4);
        v16h a;
        #pragma unroll
        for (int t = 0; t < 16; ++t) {
            u64 kj = s_key[jb + t];
            a[t] = (kj > ki) ? (_Float16)1.0f : (_Float16)0.0f;  // branchless
        }
        acc = __builtin_amdgcn_wmma_f32_16x16x32_f16(
                  false, a, false, ones, (short)0, acc, false, false);
    }

    // C/D layout: lane L, VGPR v -> M = v + 8*(L>=16), N = L%16.
    if ((tid & 15) == 0) {
        int half = tid >> 4;
        #pragma unroll
        for (int v = 0; v < 8; ++v)
            s_rank[half * 8 + v] = (int)(acc[v] + 0.5f);
    }
    __syncthreads();

    if (tid < 16) {
        int r    = rb * 16 + tid;
        int rank = s_rank[tid];
        if (rank < KTOP) {                       // sentinels rank == KTOP
            int    p  = s_id[r];
            float  sc = s_sc[r];
            size_t g  = (size_t)b * NPOS + p;
            float4 bx = ((const float4*)box)[g];
            float  kd = kindA[g];
            float* o  = out + (size_t)(b * KTOP + rank) * 6;
            o[0] = bx.x; o[1] = bx.y; o[2] = bx.z; o[3] = bx.w;
            o[4] = kd;   o[5] = sc;
        }
    }
}

// =====================================================================
// Stage D: class-aware sequential NMS per image (exact fori_loop
// semantics: suppression only from surviving i, j > i, same class),
// then final score mask keep & (score > 0). Box/kind columns untouched.
// =====================================================================
__global__ void __launch_bounds__(256) nms_kernel(float* __restrict__ out)
{
    int b   = blockIdx.x;
    int tid = threadIdx.x;               // 256 threads
    __shared__ float x1[KTOP], y1[KTOP], x2[KTOP], y2[KTOP];
    __shared__ float kd[KTOP], ar[KTOP], sc[KTOP];
    __shared__ int   keep[KTOP];

    float* ob = out + (size_t)b * KTOP * 6;
    for (int r = tid; r < KTOP; r += 256) {
        float a0 = ob[r*6+0], a1 = ob[r*6+1], a2 = ob[r*6+2], a3 = ob[r*6+3];
        x1[r] = a0; y1[r] = a1; x2[r] = a2; y2[r] = a3;
        kd[r] = ob[r*6+4];
        sc[r] = ob[r*6+5];
        ar[r] = fmaxf(a2 - a0, 0.0f) * fmaxf(a3 - a1, 0.0f);
        keep[r] = 1;
    }
    __syncthreads();

    for (int i = 0; i < KTOP; ++i) {
        if (keep[i]) {                    // uniform across block
            float xi1 = x1[i], yi1 = y1[i], xi2 = x2[i], yi2 = y2[i];
            float ai  = ar[i], ki = kd[i];
            for (int j = i + 1 + tid; j < KTOP; j += 256) {
                if (keep[j] && kd[j] == ki) {
                    float xx1 = fmaxf(xi1, x1[j]);
                    float yy1 = fmaxf(yi1, y1[j]);
                    float xx2 = fminf(xi2, x2[j]);
                    float yy2 = fminf(yi2, y2[j]);
                    float inter = fmaxf(xx2 - xx1, 0.0f) * fmaxf(yy2 - yy1, 0.0f);
                    float iou = inter / (ai + ar[j] - inter + 1e-9f);
                    if (iou > IOU_TH) keep[j] = 0;
                }
            }
        }
        __syncthreads();
    }

    for (int r = tid; r < KTOP; r += 256) {
        float s = sc[r];
        ob[r*6+5] = (keep[r] && s > 0.0f) ? s : 0.0f;
    }
}

// =====================================================================
// Host-side launcher.
// setup_inputs() dict order: cls0,cnt0,reg0, cls1,cnt1,reg1, ...
// (cnt* inputs are unused by the reference).
// =====================================================================
extern "C" void kernel_launch(void* const* d_in, const int* in_sizes, int n_in,
                              void* d_out, int out_size, void* d_ws, size_t ws_size,
                              hipStream_t stream)
{
    (void)in_sizes; (void)n_in; (void)out_size; (void)ws_size;

    const float* cls[5] = { (const float*)d_in[0], (const float*)d_in[3],
                            (const float*)d_in[6], (const float*)d_in[9],
                            (const float*)d_in[12] };
    const float* reg[5] = { (const float*)d_in[2], (const float*)d_in[5],
                            (const float*)d_in[8], (const float*)d_in[11],
                            (const float*)d_in[14] };
    float* out = (float*)d_out;

    // workspace layout (all blocks 16B-aligned: BN and CANDPAD are even)
    const size_t BN = (size_t)NB * NPOS;
    float*    ws_score = (float*)d_ws;                   // BN floats
    float*    ws_kind  = ws_score + BN;                  // BN floats
    unsigned* ws_key   = (unsigned*)(ws_kind + BN);      // BN u32
    float*    ws_box   = (float*)(ws_key + BN);          // BN*4 floats
    float*    ws_cs    = ws_box + 4 * BN;                // NB*CANDPAD floats
    int*      ws_ci    = (int*)(ws_cs + (size_t)NB * CANDPAD);   // NB*CANDPAD int
    u64*      ws_ck    = (u64*)(ws_ci + (size_t)NB * CANDPAD);   // NB*CANDPAD u64

    int total = NB * NPOS;
    decode_kernel<<<(total + 255) / 256, 256, 0, stream>>>(
        cls[0], cls[1], cls[2], cls[3], cls[4],
        reg[0], reg[1], reg[2], reg[3], reg[4],
        ws_score, ws_kind, ws_key, ws_box);

    select_topk_kernel<<<NB, 256, 0, stream>>>(ws_key, ws_score, ws_cs, ws_ci, ws_ck);

    rank_sort_kernel<<<NB * 64, 32, 0, stream>>>(ws_cs, ws_ci, ws_ck, ws_box, ws_kind, out);

    nms_kernel<<<NB, 256, 0, stream>>>(out);
}
